// BiHGT_Edge_45286135169399
// MI455X (gfx1250) — matrix operand order repaired and checked by reading, not compiled, not run on hardware
//
#include <hip/hip_runtime.h>
#include <hip/hip_bf16.h>

typedef __attribute__((ext_vector_type(16))) _Float16 v16h;
typedef __attribute__((ext_vector_type(8)))  _Float16 v8h;
typedef __attribute__((ext_vector_type(8)))  float    v8f;

#define HIDD 128

// ---------------------------------------------------------------------------
// WMMA GEMM: C[M,Ncols] = act( A[M,K] @ B[K,Ncols] + bias (+ C if accum) )
//   A  : [M,K]   f16 row-major, K multiple of 32 (zero padded)
//   BT : [Npad,K] f16  (B transposed), Npad = ceil16(Ncols), zero padded
//   M must be a multiple of 16 (true here: 50000 and 800000).
// One wave32 per 16x16 tile; 4 waves (64 output cols) per block.
// Fragment layouts per CDNA5 ISA 7.12.2 (16-bit A 16x32, B 32x16, f32 C/D).
// ---------------------------------------------------------------------------
__global__ __launch_bounds__(128)
void bh_wmma_gemm(const _Float16* __restrict__ A,
                  const _Float16* __restrict__ BT,
                  const float* __restrict__ bias,
                  float* __restrict__ Cf, _Float16* __restrict__ Ch,
                  int M, int Ncols, int K, int ldc, int relu, int accum)
{
    const int wave = threadIdx.x >> 5;
    const int lane = threadIdx.x & 31;
    const int rowA = blockIdx.x * 16;
    const int colB = (blockIdx.y * 4 + wave) * 16;
    const int Npad = (Ncols + 15) & ~15;
    if (colB >= Npad) return;

    const int m  = rowA + (lane & 15);
    const int kA = (lane >> 4) * 8;    // A: lanes16-31 start at K+8
    const int n  = colB + (lane & 15);
    const int kB = (lane >> 4) * 8;    // B: lanes16-31 start at K+8

    const _Float16* pa = A  + (size_t)m * K + kA;
    const _Float16* pb = BT + (size_t)n * K + kB;

    v8f c = {};
    for (int k0 = 0; k0 < K; k0 += 32) {
        v8h alo = *(const v8h*)(pa + k0);        // K = k0+kA .. +7   (vgpr 0..3)
        v8h ahi = *(const v8h*)(pa + k0 + 16);   // K = k0+16+kA .. +7 (vgpr 4..7)
        v8h blo = *(const v8h*)(pb + k0);        // K = k0+kB .. +7
        v8h bhi = *(const v8h*)(pb + k0 + 16);   // K = k0+16+kB .. +7
        v16h av, bvv;
#pragma unroll
        for (int i = 0; i < 8; ++i) {
            av[i] = alo[i]; av[i + 8] = ahi[i];
            bvv[i] = blo[i]; bvv[i + 8] = bhi[i];
        }
        c = __builtin_amdgcn_wmma_f32_16x16x32_f16(
                /*neg_a=*/false, av, /*neg_b=*/false, bvv,
                /*c_mod=*/(short)0, c, /*reuse_a=*/false, /*reuse_b=*/false);
    }

    if (n < Ncols) {
        const float bsv = bias ? bias[n] : 0.0f;
        const int mrow = rowA + (lane >> 4) * 8; // C/D: lanes16-31 hold M=8..15
#pragma unroll
        for (int r = 0; r < 8; ++r) {
            float val = c[r] + bsv;
            size_t off = (size_t)(mrow + r) * ldc + n;
            if (accum && Cf) val += Cf[off];
            if (relu) val = val > 0.f ? val : 0.f;
            if (Cf) Cf[off] = val;
            if (Ch) Ch[off] = (_Float16)val;
        }
    }
}

// ---- weight convert+transpose+pad: W[K,N] f32 -> WT[Npad,Kpad] f16 --------
__global__ void bh_conv_transpose(const float* __restrict__ W, _Float16* __restrict__ WT,
                                  int K, int N, int Kpad, int Npad)
{
    int idx = blockIdx.x * blockDim.x + threadIdx.x;
    int total = Npad * Kpad;
    if (idx >= total) return;
    int n = idx / Kpad, k = idx - n * Kpad;
    float v = (n < N && k < K) ? W[(size_t)k * N + n] : 0.f;
    WT[idx] = (_Float16)v;
}

// ---- activation convert+pad: X[M,K] f32 -> X16[M,Kpad] f16 ----------------
__global__ void bh_conv_f16(const float* __restrict__ X, _Float16* __restrict__ X16,
                            int M, int K, int Kpad)
{
    int idx = blockIdx.x * blockDim.x + threadIdx.x;
    if (idx >= M * Kpad) return;
    int m = idx / Kpad, k = idx - m * Kpad;
    X16[idx] = (k < K) ? (_Float16)X[(size_t)m * K + k] : (_Float16)0.f;
}

// ---- edge encoding: [sin(dt*w), cos(dt*w), w_norm, infl, cos_sim, 0-pad] --
__global__ void bh_edge_enc(const float* __restrict__ ea, const float* __restrict__ tew,
                            _Float16* __restrict__ enc, int E)
{
    int idx = blockIdx.x * blockDim.x + threadIdx.x;
    if (idx >= E * 96) return;
    int e = idx / 96, j = idx - e * 96;
    float v;
    if (j < 32)       v = sinf(ea[e * 4 + 1] * tew[j]);
    else if (j < 64)  v = cosf(ea[e * 4 + 1] * tew[j - 32]);
    else if (j == 64) v = ea[e * 4 + 0];
    else if (j == 65) v = ea[e * 4 + 2];
    else if (j == 66) v = ea[e * 4 + 3];
    else              v = 0.f;
    enc[idx] = (_Float16)v;
}

__global__ void bh_fill(float* __restrict__ p, float v, size_t n)
{
    size_t i = (size_t)blockIdx.x * blockDim.x + threadIdx.x;
    size_t stride = (size_t)gridDim.x * blockDim.x;
    for (; i < n; i += stride) p[i] = v;
}

// ---- attention logits + segment max (ordered-uint atomicMax trick) --------
__global__ void bh_alpha(const int* __restrict__ src, const int* __restrict__ dst,
                         const float* __restrict__ q, const float* __restrict__ k,
                         const _Float16* __restrict__ e16,
                         float* __restrict__ alpha, unsigned* __restrict__ amax, int E)
{
    int idx = blockIdx.x * blockDim.x + threadIdx.x;
    if (idx >= E * 4) return;
    int e = idx >> 2, hh = idx & 3;
    int s = src[e], d = dst[e];
    const float*    qd = q  + (size_t)d * HIDD + hh * 32;
    const float*    ks = k  + (size_t)s * HIDD + hh * 32;
    const _Float16* ee = e16 + (size_t)e * HIDD + hh * 32;
    float a = 0.f;
#pragma unroll
    for (int c = 0; c < 32; ++c) a += qd[c] * (ks[c] + (float)ee[c]);
    a *= 0.17677669529663687f;  // 1/sqrt(32)
    alpha[idx] = a;
    unsigned u = __float_as_uint(a);
    u = (u >> 31) ? ~u : (u | 0x80000000u);
    atomicMax(&amax[(size_t)d * 4 + hh], u);
}

__global__ void bh_expden(const int* __restrict__ dst, float* __restrict__ alpha,
                          const unsigned* __restrict__ amax, float* __restrict__ den, int E)
{
    int idx = blockIdx.x * blockDim.x + threadIdx.x;
    if (idx >= E * 4) return;
    int e = idx >> 2, hh = idx & 3;
    int d = dst[e];
    unsigned u = amax[(size_t)d * 4 + hh];
    u = (u >> 31) ? (u & 0x7fffffffu) : ~u;
    float mx = __uint_as_float(u);
    float ex = expf(alpha[idx] - mx);
    alpha[idx] = ex;
    atomicAdd(&den[(size_t)d * 4 + hh], ex);
}

__global__ void bh_aggregate(const int* __restrict__ src, const int* __restrict__ dst,
                             const float* __restrict__ ex, const float* __restrict__ den,
                             const float* __restrict__ v, const _Float16* __restrict__ e16,
                             float* __restrict__ acc, int E)
{
    int idx = blockIdx.x * blockDim.x + threadIdx.x;
    if (idx >= E * 4) return;
    int e = idx >> 2, hh = idx & 3;
    int s = src[e], d = dst[e];
    float a = ex[idx] / (den[(size_t)d * 4 + hh] + 1e-16f);
    const float*    vs = v   + (size_t)s * HIDD + hh * 32;
    const _Float16* ee = e16 + (size_t)e * HIDD + hh * 32;
    float*          ad = acc + (size_t)d * HIDD + hh * 32;
#pragma unroll
    for (int c = 0; c < 32; ++c) atomicAdd(&ad[c], a * (vs[c] + (float)ee[c]));
}

// ---- fused ReLU + LayerNorm, one wave32 per node --------------------------
__global__ __launch_bounds__(256)
void bh_relu_ln(const float* __restrict__ acc, const float* __restrict__ gamma,
                const float* __restrict__ beta, float* __restrict__ hout,
                _Float16* __restrict__ h16, int Nn)
{
    int wave = threadIdx.x >> 5, lane = threadIdx.x & 31;
    int node = blockIdx.x * 8 + wave;
    if (node >= Nn) return;
    float t[4]; float sum = 0.f;
#pragma unroll
    for (int i = 0; i < 4; ++i) {
        float x = acc[(size_t)node * HIDD + lane * 4 + i];
        x = x > 0.f ? x : 0.f;
        t[i] = x; sum += x;
    }
#pragma unroll
    for (int off = 16; off > 0; off >>= 1) sum += __shfl_xor(sum, off, 32);
    float mu = sum * (1.f / 128.f);
    float var = 0.f;
#pragma unroll
    for (int i = 0; i < 4; ++i) { float dd = t[i] - mu; var += dd * dd; }
#pragma unroll
    for (int off = 16; off > 0; off >>= 1) var += __shfl_xor(var, off, 32);
    var *= (1.f / 128.f);
    float inv = rsqrtf(var + 1e-5f);
#pragma unroll
    for (int i = 0; i < 4; ++i) {
        int c = lane * 4 + i;
        float y = (t[i] - mu) * inv * gamma[c] + beta[c];
        hout[(size_t)node * HIDD + c] = y;
        h16 [(size_t)node * HIDD + c] = (_Float16)y;
    }
}

// ---- JumpingKnowledge head: out[n,4] = [h0|h1|h2] @ W_out + b_out ---------
__global__ void bh_out_head(const float* __restrict__ h0, const float* __restrict__ h1,
                            const float* __restrict__ h2, const float* __restrict__ Wout,
                            const float* __restrict__ bout, float* __restrict__ out, int Nn)
{
    int idx = blockIdx.x * blockDim.x + threadIdx.x;
    if (idx >= Nn * 4) return;
    int nd = idx >> 2, o = idx & 3;
    float a = bout[o];
    const float* r0 = h0 + (size_t)nd * HIDD;
    const float* r1 = h1 + (size_t)nd * HIDD;
    const float* r2 = h2 + (size_t)nd * HIDD;
    for (int c = 0; c < HIDD; ++c) a += r0[c] * Wout[c * 4 + o];
    for (int c = 0; c < HIDD; ++c) a += r1[c] * Wout[(HIDD + c) * 4 + o];
    for (int c = 0; c < HIDD; ++c) a += r2[c] * Wout[(2 * HIDD + c) * 4 + o];
    out[idx] = a;
}

__global__ void bh_copy(const float* __restrict__ s, float* __restrict__ d, size_t n)
{
    size_t i = (size_t)blockIdx.x * blockDim.x + threadIdx.x;
    if (i < n) d[i] = s[i];
}

// ===========================================================================
extern "C" void kernel_launch(void* const* d_in, const int* in_sizes, int n_in,
                              void* d_out, int out_size, void* d_ws, size_t ws_size,
                              hipStream_t stream)
{
    const float* x     = (const float*)d_in[0];
    const int*   ei_td = (const int*)  d_in[1];
    const int*   ei_bu = (const int*)  d_in[2];
    const float* ea_td = (const float*)d_in[3];
    const float* ea_bu = (const float*)d_in[4];
    const float* W_in  = (const float*)d_in[5];
    const float* b_in  = (const float*)d_in[6];
    const float* te_w  = (const float*)d_in[7];
    const float* Wq    = (const float*)d_in[8];
    const float* bq    = (const float*)d_in[9];
    const float* Wk    = (const float*)d_in[10];
    const float* bk    = (const float*)d_in[11];
    const float* Wv    = (const float*)d_in[12];
    const float* bv    = (const float*)d_in[13];
    const float* We    = (const float*)d_in[14];
    const float* Wskip = (const float*)d_in[15];
    const float* bskip = (const float*)d_in[16];
    const float* gamma = (const float*)d_in[17];
    const float* beta  = (const float*)d_in[18];
    const float* W_out = (const float*)d_in[19];
    const float* b_out = (const float*)d_in[20];
    const float* W_dec = (const float*)d_in[21];
    const float* b_dec = (const float*)d_in[22];

    const int N = in_sizes[0] / 300;       // 50000
    const int E = in_sizes[1] / 2;         // 800000
    const int KIN = 300, KINP = 320, ENCP = 96, NDEC = 300, NDECP = 304;

    // ---- workspace carve-out (everything rewritten every call) ----
    char* wp = (char*)d_ws;
    auto carve = [&](size_t bytes) -> void* {
        void* p = wp; wp += (bytes + 255) & ~(size_t)255; return p;
    };
    _Float16* x16   = (_Float16*)carve((size_t)N * KINP * 2);
    _Float16* h16   = (_Float16*)carve((size_t)N * HIDD * 2);
    float*    h0    = (float*)   carve((size_t)N * HIDD * 4);
    float*    h1    = (float*)   carve((size_t)N * HIDD * 4);
    float*    h2    = (float*)   carve((size_t)N * HIDD * 4);
    float*    qb    = (float*)   carve((size_t)N * HIDD * 4);
    float*    kb    = (float*)   carve((size_t)N * HIDD * 4);
    float*    vb    = (float*)   carve((size_t)N * HIDD * 4);
    float*    acc   = (float*)   carve((size_t)N * HIDD * 4);
    _Float16* enc16 = (_Float16*)carve((size_t)E * ENCP * 2);
    _Float16* e16   = (_Float16*)carve((size_t)E * HIDD * 2);
    float*    alpha = (float*)   carve((size_t)E * 4 * 4);
    unsigned* amax  = (unsigned*)carve((size_t)N * 4 * 4);
    float*    den   = (float*)   carve((size_t)N * 4 * 4);
    _Float16* WinT  = (_Float16*)carve((size_t)HIDD * KINP * 2);
    _Float16* WqT   = (_Float16*)carve((size_t)4 * HIDD * HIDD * 2);
    _Float16* WkT   = (_Float16*)carve((size_t)4 * HIDD * HIDD * 2);
    _Float16* WvT   = (_Float16*)carve((size_t)4 * HIDD * HIDD * 2);
    _Float16* WsT   = (_Float16*)carve((size_t)4 * HIDD * HIDD * 2);
    _Float16* WeT   = (_Float16*)carve((size_t)4 * HIDD * ENCP * 2);
    _Float16* WdT   = (_Float16*)carve((size_t)NDECP * HIDD * 2);

    const int TB = 256;
    auto blocks = [](long long n, int tb) { return (unsigned)((n + tb - 1) / tb); };

    // ---- weight + input conversion ----
    bh_conv_f16<<<blocks((long long)N * KINP, TB), TB, 0, stream>>>(x, x16, N, KIN, KINP);
    bh_conv_transpose<<<blocks(HIDD * KINP, TB), TB, 0, stream>>>(W_in, WinT, KIN, HIDD, KINP, HIDD);
    for (int s = 0; s < 4; ++s) {
        bh_conv_transpose<<<blocks(HIDD * HIDD, TB), TB, 0, stream>>>(
            Wq + (size_t)s * HIDD * HIDD, WqT + (size_t)s * HIDD * HIDD, HIDD, HIDD, HIDD, HIDD);
        bh_conv_transpose<<<blocks(HIDD * HIDD, TB), TB, 0, stream>>>(
            Wk + (size_t)s * HIDD * HIDD, WkT + (size_t)s * HIDD * HIDD, HIDD, HIDD, HIDD, HIDD);
        bh_conv_transpose<<<blocks(HIDD * HIDD, TB), TB, 0, stream>>>(
            Wv + (size_t)s * HIDD * HIDD, WvT + (size_t)s * HIDD * HIDD, HIDD, HIDD, HIDD, HIDD);
        bh_conv_transpose<<<blocks(HIDD * HIDD, TB), TB, 0, stream>>>(
            Wskip + (size_t)s * HIDD * HIDD, WsT + (size_t)s * HIDD * HIDD, HIDD, HIDD, HIDD, HIDD);
        bh_conv_transpose<<<blocks(HIDD * ENCP, TB), TB, 0, stream>>>(
            We + (size_t)s * 67 * HIDD, WeT + (size_t)s * HIDD * ENCP, 67, HIDD, ENCP, HIDD);
    }
    bh_conv_transpose<<<blocks(NDECP * HIDD, TB), TB, 0, stream>>>(W_dec, WdT, HIDD, NDEC, HIDD, NDECP);

    // ---- h0 = relu(x @ W_in + b_in), also f16 copy ----
    bh_wmma_gemm<<<dim3(N / 16, 2), 128, 0, stream>>>(
        x16, WinT, b_in, h0, h16, N, HIDD, KINP, HIDD, /*relu=*/1, /*accum=*/0);

    float* houts[3] = { h0, h1, h2 };
    for (int layer = 0; layer < 2; ++layer) {
        for (int et = 0; et < 2; ++et) {
            const int s = layer * 2 + et;
            const int* src = (et == 0) ? ei_td : ei_bu;
            const int* dst = src + E;
            const float* ea = (et == 0) ? ea_td : ea_bu;

            // q, k, v
            bh_wmma_gemm<<<dim3(N / 16, 2), 128, 0, stream>>>(
                h16, WqT + (size_t)s * HIDD * HIDD, bq + (size_t)s * HIDD,
                qb, (_Float16*)nullptr, N, HIDD, HIDD, HIDD, 0, 0);
            bh_wmma_gemm<<<dim3(N / 16, 2), 128, 0, stream>>>(
                h16, WkT + (size_t)s * HIDD * HIDD, bk + (size_t)s * HIDD,
                kb, (_Float16*)nullptr, N, HIDD, HIDD, HIDD, 0, 0);
            bh_wmma_gemm<<<dim3(N / 16, 2), 128, 0, stream>>>(
                h16, WvT + (size_t)s * HIDD * HIDD, bv + (size_t)s * HIDD,
                vb, (_Float16*)nullptr, N, HIDD, HIDD, HIDD, 0, 0);
            // skip into accumulator (write for TD, accumulate for BU)
            bh_wmma_gemm<<<dim3(N / 16, 2), 128, 0, stream>>>(
                h16, WsT + (size_t)s * HIDD * HIDD, bskip + (size_t)s * HIDD,
                acc, (_Float16*)nullptr, N, HIDD, HIDD, HIDD, 0, /*accum=*/et);

            // per-edge key/value offsets e = enc @ We  (f16 output)
            bh_edge_enc<<<blocks((long long)E * ENCP, TB), TB, 0, stream>>>(ea, te_w, enc16, E);
            bh_wmma_gemm<<<dim3(E / 16, 2), 128, 0, stream>>>(
                enc16, WeT + (size_t)s * HIDD * ENCP, (const float*)nullptr,
                (float*)nullptr, e16, E, HIDD, ENCP, HIDD, 0, 0);

            // scatter-softmax attention
            bh_fill<<<1024, TB, 0, stream>>>((float*)amax, 0.f, (size_t)N * 4);
            bh_fill<<<1024, TB, 0, stream>>>(den, 0.f, (size_t)N * 4);
            bh_alpha<<<blocks((long long)E * 4, TB), TB, 0, stream>>>(
                src, dst, qb, kb, e16, alpha, amax, E);
            bh_expden<<<blocks((long long)E * 4, TB), TB, 0, stream>>>(dst, alpha, amax, den, E);
            bh_aggregate<<<blocks((long long)E * 4, TB), TB, 0, stream>>>(
                src, dst, alpha, den, vb, e16, acc, E);
        }
        // h_{l+1} = LN(relu(acc)) * gamma + beta ; refresh f16 copy
        bh_relu_ln<<<(N + 7) / 8, 256, 0, stream>>>(
            acc, gamma + (size_t)layer * HIDD, beta + (size_t)layer * HIDD,
            houts[layer + 1], h16, N);
    }

    // ---- outputs: [out | z | x_rec] flat in d_out ----
    float* out_p = (float*)d_out;
    float* z_p   = out_p + (size_t)N * 4;
    float* xr_p  = z_p + (size_t)N * HIDD;
    bh_out_head<<<blocks((long long)N * 4, TB), TB, 0, stream>>>(h0, h1, h2, W_out, b_out, out_p, N);
    bh_copy<<<blocks((long long)N * HIDD, TB), TB, 0, stream>>>(h2, z_p, (size_t)N * HIDD);
    bh_wmma_gemm<<<dim3(N / 16, (NDECP + 63) / 64), 128, 0, stream>>>(
        h16, WdT, b_dec, xr_p, (_Float16*)nullptr, N, NDEC, HIDD, NDEC, 0, 0);
}
